// Sequence_1022202217222
// MI455X (gfx1250) — matrix operand order repaired and checked by reading, not compiled, not run on hardware
//
#include <hip/hip_runtime.h>

typedef __attribute__((ext_vector_type(16))) _Float16 v16h;
typedef __attribute__((ext_vector_type(8)))  _Float16 v8h;
typedef __attribute__((ext_vector_type(8)))  float    v8f;

#define T_STEPS 1024
#define BATCH   256
#define HID     120
#define NLAYER  4
#define GATES   480        // 4*HID
#define KPAD    256        // K padded: [inp | h(120) | 1.0 @240 | zeros]
#define NTILES  30         // GATES/16 column tiles
#define MAIN_THREADS 960   // exactly 30 wave32s: one gate column-tile per wave

// wpack: f16, layout [l][n][s][lane][e]; strides e:1 lane:16 s:512 n:4096 l:122880
#define WPACK_ELEMS (NLAYER * NTILES * 8 * 32 * 16)   // 491520 f16 = 983040 B

// ---------------------------------------------------------------------------
// Prep: repack weights into a WMMA B-operand lane layout with contiguous k (16h + e), f16.
// Wcat[l][K][g]: l==0: K=0 -> W_ih0[g]; 1..120 -> W_hh[0][g][K-1]
//               l>=1: K<120 -> W_ih[l-1][g][K]; K<240 -> W_hh[l][g][K-120]
// All layers:   K==240 -> b_ih[l][g]+b_hh[l][g]   (bias folded into GEMM,
//               matched by a constant 1.0 planted in A column 240)
// B element e of lane t (k-step s) = Wcat[32*s + 16*(t>>4) + e][16*n + (t&15)]
// ---------------------------------------------------------------------------
__global__ void prep_kernel(const float* __restrict__ W_ih0,
                            const float* __restrict__ W_ih,
                            const float* __restrict__ W_hh,
                            const float* __restrict__ b_ih,
                            const float* __restrict__ b_hh,
                            _Float16* __restrict__ wpack)
{
    int q = blockIdx.x * blockDim.x + threadIdx.x;
    if (q >= WPACK_ELEMS) return;
    int l = q / 122880; int r = q % 122880;
    int n = r / 4096;   r %= 4096;
    int s = r / 512;    r %= 512;
    int t = r / 16;     int e = r % 16;
    int g = 16 * n + (t & 15);
    int K = 32 * s + 16 * (t >> 4) + e;
    float w = 0.0f;
    if (K == 240) {
        w = b_ih[l * GATES + g] + b_hh[l * GATES + g];
    } else if (l == 0) {
        if (K == 0)        w = W_ih0[g];
        else if (K <= HID) w = W_hh[(size_t)g * HID + (K - 1)];
    } else {
        if (K < HID)          w = W_ih[((size_t)(l - 1) * GATES + g) * HID + K];
        else if (K < 2 * HID) w = W_hh[((size_t)l * GATES + g) * HID + (K - HID)];
    }
    wpack[q] = (_Float16)w;
}

// Branch-free activations: v_exp_f32 + v_rcp_f32, correct saturation at +-inf.
__device__ __forceinline__ float sigm_fast(float v) {
    return __builtin_amdgcn_rcpf(1.0f + __expf(-v));
}
__device__ __forceinline__ float tanh_fast(float v) {
    return 1.0f - 2.0f * __builtin_amdgcn_rcpf(__expf(2.0f * v) + 1.0f);
}

// ---------------------------------------------------------------------------
// Persistent kernel: one workgroup per 16-row batch slice, loops all T steps.
// Layer loop deliberately NOT unrolled: keeps exactly one layer's B tiles
// (64 VGPRs) live, preserving the staggered-wait WMMA pipeline without
// scratch spills.
// ---------------------------------------------------------------------------
__global__ void __launch_bounds__(MAIN_THREADS, 1)
lstm_persistent(const float* __restrict__ x,
                const float* __restrict__ h0,
                const float* __restrict__ c0,
                const float* __restrict__ W_lin,
                const float* __restrict__ b_lin,
                const _Float16* __restrict__ wpack,
                float* __restrict__ out)
{
    extern __shared__ char smem[];
    _Float16* Abuf   = (_Float16*)smem;                             // [4][16][256] f16, 32768 B
    float*    gatesS = (float*)(smem + NLAYER * 16 * KPAD * 2);     // [16][480] f32, 30720 B
    float*    wlS    = gatesS + 16 * GATES;                         // [120] f32, 480 B

    const int tid   = threadIdx.x;
    const int b0    = blockIdx.x * 16;     // batch slice base
    const int lane  = tid & 31;
    const int wave  = tid >> 5;            // == column tile n, 0..29
    const int nrow  = lane & 15;           // A row index / D column index
    const int khalf = lane >> 4;

    for (int i = tid; i < NLAYER * 16 * KPAD; i += MAIN_THREADS) Abuf[i] = (_Float16)0.0f;
    for (int i = tid; i < HID; i += MAIN_THREADS) wlS[i] = W_lin[i];
    __syncthreads();

    // elementwise mapping (loop-invariant): thread covers e = tid and tid+960
    const int m0 = tid / HID,                  j0 = tid % HID;
    const int m1 = (tid + MAIN_THREADS) / HID, j1 = (tid + MAIN_THREADS) % HID;

    // c-state in registers (2 elements per thread per layer)
    float creg[NLAYER * 2];
    #pragma unroll
    for (int l = 0; l < NLAYER; ++l) {
        #pragma unroll
        for (int k = 0; k < 2; ++k) {
            int m = k ? m1 : m0, j = k ? j1 : j0;
            int gi = ((l * BATCH) + b0 + m) * HID + j;
            creg[l * 2 + k] = c0[gi];
            int col = (l == 0) ? (1 + j) : (HID + j);
            Abuf[(l * 16 + m) * KPAD + col] = (_Float16)h0[gi];
        }
    }
    // constant-1 column matching the bias row (K=240) of wpack
    for (int i = tid; i < NLAYER * 16; i += MAIN_THREADS)
        Abuf[i * KPAD + 240] = (_Float16)1.0f;
    // x for t=0
    if (tid < 16) Abuf[tid * KPAD] = (_Float16)x[b0 + tid];
    __syncthreads();

    const float blin = b_lin[0];

    for (int t = 0; t < T_STEPS; ++t) {
        // prefetch next step's x into a register; latency hidden by 4 layers
        float xnext = 0.0f;
        if (tid < 16 && t + 1 < T_STEPS) xnext = x[(t + 1) * BATCH + b0 + tid];

        #pragma unroll 1   // rolled: avoids cross-layer B-load hoisting/spills
        for (int l = 0; l < NLAYER; ++l) {
            // ---- GEMM: wave n -> gates columns [16n,16n+16), K=256 (8 WMMAs);
            //      all 30 waves participate: no divergence, no exec masking ----
            {
                v8f acc = {0.f, 0.f, 0.f, 0.f, 0.f, 0.f, 0.f, 0.f};
                const _Float16* Arow = Abuf + (l * 16 + nrow) * KPAD;
                const v16h* wp = (const v16h*)wpack + ((l * NTILES + wave) * 8) * 32 + lane;
                #pragma unroll
                for (int s = 0; s < 8; ++s) {
                    v8h alo = *(const v8h*)(Arow + 32 * s + 16 * khalf);
                    v8h ahi = *(const v8h*)(Arow + 32 * s + 16 * khalf + 8);
                    v16h a  = __builtin_shufflevector(alo, ahi,
                                0,1,2,3,4,5,6,7,8,9,10,11,12,13,14,15);
                    v16h bm = wp[s * 32];   // pre-packed B operand, 32 B/lane from L2
                    acc = __builtin_amdgcn_wmma_f32_16x16x32_f16(
                              false, a, false, bm, (short)0, acc, false, false);
                }
                // D layout: lane t, VGPR r holds D[r + 8*(t>>4)][t&15]
                #pragma unroll
                for (int r = 0; r < 8; ++r)
                    gatesS[(r + 8 * khalf) * GATES + 16 * wave + nrow] = acc[r];
            }
            __syncthreads();

            // ---- elementwise LSTM cell update (bias already inside gates) ----
            #pragma unroll
            for (int k = 0; k < 2; ++k) {
                const int m = k ? m1 : m0, j = k ? j1 : j0;
                const float* gr = gatesS + m * GATES;
                float ig = gr[j];
                float fg = gr[HID + j];
                float gg = gr[2 * HID + j];
                float og = gr[3 * HID + j];
                float cn = sigm_fast(fg) * creg[l * 2 + k]
                         + sigm_fast(ig) * tanh_fast(gg);
                creg[l * 2 + k] = cn;
                float hn = sigm_fast(og) * tanh_fast(cn);
                _Float16 hh = (_Float16)hn;
                int colr = (l == 0) ? (1 + j) : (HID + j);     // recurrent slot
                Abuf[(l * 16 + m) * KPAD + colr] = hh;
                if (l < NLAYER - 1)                            // next layer's input
                    Abuf[((l + 1) * 16 + m) * KPAD + j] = hh;
            }
            // stage next step's x while we're between barriers anyway
            if (l == NLAYER - 1 && tid < 16 && t + 1 < T_STEPS)
                Abuf[tid * KPAD] = (_Float16)xnext;
            __syncthreads();
        }

        // ---- out[t,b] = h3 . W_lin + b_lin : 16 waves, lane-parallel dot ----
        // (overlaps next iteration's layer-0 GEMM; readers only, and A_3's
        //  recurrent slots aren't rewritten until l==3 of step t+1)
        if (wave < 16) {
            const _Float16* hp = Abuf + (3 * 16 + wave) * KPAD + HID;
            float p = 0.0f;
            for (int j = lane; j < HID; j += 32) p += (float)hp[j] * wlS[j];
            #pragma unroll
            for (int off = 16; off > 0; off >>= 1)
                p += __shfl_xor(p, off, 32);
            if (lane == 0) out[t * BATCH + b0 + wave] = p + blin;
        }
    }
}

extern "C" void kernel_launch(void* const* d_in, const int* in_sizes, int n_in,
                              void* d_out, int out_size, void* d_ws, size_t ws_size,
                              hipStream_t stream)
{
    (void)in_sizes; (void)n_in; (void)out_size; (void)ws_size;
    const float* x     = (const float*)d_in[0];
    const float* h0    = (const float*)d_in[1];
    const float* c0    = (const float*)d_in[2];
    const float* W_ih0 = (const float*)d_in[3];
    const float* W_ih  = (const float*)d_in[4];
    const float* W_hh  = (const float*)d_in[5];
    const float* b_ih  = (const float*)d_in[6];
    const float* b_hh  = (const float*)d_in[7];
    const float* W_lin = (const float*)d_in[8];
    const float* b_lin = (const float*)d_in[9];
    float* out = (float*)d_out;

    _Float16* wpack = (_Float16*)d_ws;

    prep_kernel<<<(WPACK_ELEMS + 255) / 256, 256, 0, stream>>>(
        W_ih0, W_ih, W_hh, b_ih, b_hh, wpack);

    size_t smem = (size_t)NLAYER * 16 * KPAD * 2 + 16 * GATES * 4 + HID * 4; // 63968 B
    lstm_persistent<<<BATCH / 16, MAIN_THREADS, smem, stream>>>(
        x, h0, c0, W_lin, b_lin, wpack, out);
}
